// EdgeConv2d_6150393168688
// MI455X (gfx1250) — matrix operand order repaired and checked by reading, not compiled, not run on hardware
//
#include <hip/hip_runtime.h>
#include <cstdint>

#define B_    2
#define C_    64
#define N_    32768
#define K_    16
#define OUT_  64
#define EPS_  1e-5f

typedef __attribute__((ext_vector_type(16))) _Float16 v16h;
typedef __attribute__((ext_vector_type(8)))  _Float16 v8h;
typedef __attribute__((ext_vector_type(8)))  float    v8f;

union V16 { v16h v; v8h h[2]; };

// ---------------- prep W: Wa = W[:, :64]-W[:,64:], Wb = W[:,64:]  (f32 -> f16)
__global__ void prep_w(const float* __restrict__ W,
                       _Float16* __restrict__ Wa, _Float16* __restrict__ Wb) {
    int g = blockIdx.x * blockDim.x + threadIdx.x;     // 0..4095
    if (g >= OUT_ * 64) return;
    int o = g >> 6, c = g & 63;
    float w1 = W[o * 128 + c];
    float w2 = W[o * 128 + 64 + c];
    Wa[g] = (_Float16)(w1 - w2);
    Wb[g] = (_Float16)w2;
}

// ---------------- transpose x [B,C,N] f32 -> xT [B,N,64] f16 (LDS tiled)
__global__ void prep_x(const float* __restrict__ x, _Float16* __restrict__ xT) {
    __shared__ _Float16 t[64][65];
    int b  = blockIdx.x >> 9;               // 512 blocks per batch
    int n0 = (blockIdx.x & 511) << 6;
    #pragma unroll
    for (int i = 0; i < 16; ++i) {
        int id = threadIdx.x + (i << 8);    // 0..4095
        int c = id >> 6, n = id & 63;
        t[c][n] = (_Float16)x[((b * 64 + c) << 15) + n0 + n];   // coalesced read
    }
    __syncthreads();
    #pragma unroll
    for (int i = 0; i < 16; ++i) {
        int id = threadIdx.x + (i << 8);
        int n = id >> 6, c = id & 63;
        xT[(((b << 15) + n0 + n) << 6) + c] = t[c][n];          // coalesced write
    }
}

// ---------------- main: per node (b,n): y[64,16] = Wa@Xi + Wb@Xj via WMMA,
// then max/min over k, and per-channel sum/sumsq accumulation.
__global__ void __launch_bounds__(256, 1)
edge_main(const _Float16* __restrict__ xT,
          const _Float16* __restrict__ Wa,
          const _Float16* __restrict__ Wb,
          const int* __restrict__ eidx,
          float* __restrict__ ymax,
          float* __restrict__ ymin,
          float* __restrict__ partial) {
    __shared__ float ytile[8][16][68];   // [wave][k][o], padded (bank-conflict-free)
    __shared__ float bacc[128];          // [0..63]=sum, [64..127]=sumsq
    const int tid  = threadIdx.x;
    const int w    = tid >> 5;
    const int lane = tid & 31;
    const int lo16 = lane & 15;
    const int hi   = lane >> 4;

    if (tid < 128) bacc[tid] = 0.f;
    __syncthreads();

    // Preload 16 A-fragments (loop invariant). 16-bit A layout used here (contiguous k):
    // lane M = lane&15;  element j -> c = j + 16*(lane>>4) (+32*h chunk)
    V16 Afrag[2][4][2];
    #pragma unroll
    for (int m = 0; m < 2; ++m) {
        const _Float16* Wm = m ? Wb : Wa;
        #pragma unroll
        for (int t = 0; t < 4; ++t)
            #pragma unroll
            for (int h = 0; h < 2; ++h) {
                const _Float16* p = Wm + ((t * 16 + lo16) * 64 + h * 32 + hi * 16);
                Afrag[m][t][h].h[0] = *(const v8h*)p;
                Afrag[m][t][h].h[1] = *(const v8h*)(p + 8);
            }
    }

    const int gw     = blockIdx.x * 8 + w;
    const int nwaves = gridDim.x * 8;

    for (int node = gw; node < B_ * N_; node += nwaves) {
        const int b = node >> 15;
        const int n = node & (N_ - 1);
        // edge_index shape (2,B,N,K): row1 = center idx, row0 = neighbor idx
        const int ic = eidx[((B_ + b) * N_ + n) * K_ + lo16];
        const int ij = eidx[((     b) * N_ + n) * K_ + lo16];
        // B-frag layout used here (contiguous k): lane = k column (lane&15); lane-half selects c 0..15 / 16..31
        // -> each lane loads 32 contiguous bytes from the (transposed) feature column.
        const _Float16* pI = xT + ((((b << 15) + ic) << 6) + hi * 16);
        const _Float16* pJ = xT + ((((b << 15) + ij) << 6) + hi * 16);
        V16 bi0, bi1, bj0, bj1;
        bi0.v = *(const v16h*)pI;          // c 0..31
        bi1.v = *(const v16h*)(pI + 32);   // c 32..63
        bj0.v = *(const v16h*)pJ;
        bj1.v = *(const v16h*)(pJ + 32);

        #pragma unroll
        for (int t = 0; t < 4; ++t) {
            v8f acc = {};
            acc = __builtin_amdgcn_wmma_f32_16x16x32_f16(false, Afrag[0][t][0].v, false, bi0.v, (short)0, acc, false, false);
            acc = __builtin_amdgcn_wmma_f32_16x16x32_f16(false, Afrag[0][t][1].v, false, bi1.v, (short)0, acc, false, false);
            acc = __builtin_amdgcn_wmma_f32_16x16x32_f16(false, Afrag[1][t][0].v, false, bj0.v, (short)0, acc, false, false);
            acc = __builtin_amdgcn_wmma_f32_16x16x32_f16(false, Afrag[1][t][1].v, false, bj1.v, (short)0, acc, false, false);
            // D layout: VGPR r -> M = r + 8*hi, col = lane&15 -> o = 16t + r + 8*hi
            float* dst = &ytile[w][lo16][t * 16 + hi * 8];
            *(float4*)dst       = make_float4(acc[0], acc[1], acc[2], acc[3]);
            *((float4*)dst + 1) = make_float4(acc[4], acc[5], acc[6], acc[7]);
        }
        __builtin_amdgcn_wave_barrier();   // in-wave LDS is in-order; keep sched order

        #pragma unroll
        for (int ch = 0; ch < 2; ++ch) {
            const int o = lane + ch * 32;
            float s = 0.f, sq = 0.f, mx = -3.4e38f, mn = 3.4e38f;
            #pragma unroll
            for (int kk = 0; kk < 16; ++kk) {
                float v = ytile[w][kk][o];
                s += v;
                sq = fmaf(v, v, sq);
                mx = fmaxf(mx, v);
                mn = fminf(mn, v);
            }
            const int gb = (node << 6) + o;   // [b][n][o] layout: coalesced
            ymax[gb] = mx;
            ymin[gb] = mn;
            atomicAdd(&bacc[o], s);           // ds_add_f32
            atomicAdd(&bacc[64 + o], sq);
        }
        __builtin_amdgcn_wave_barrier();
    }
    __syncthreads();
    if (tid < 128) partial[blockIdx.x * 128 + tid] = bacc[tid];
}

// ---------------- stats: mean/var -> per-channel affine (a, shift)
__global__ void stats_k(const float* __restrict__ partial,
                        const float* __restrict__ gamma,
                        const float* __restrict__ beta,
                        float* __restrict__ ab, int nblocks) {
    int o = threadIdx.x;
    if (o >= 64) return;
    float s = 0.f, sq = 0.f;
    for (int bl = 0; bl < nblocks; ++bl) {
        s  += partial[bl * 128 + o];
        sq += partial[bl * 128 + 64 + o];
    }
    const float inv = 1.f / (float)(B_ * N_ * K_);
    float mean = s * inv;
    float var  = fmaxf(sq * inv - mean * mean, 0.f);
    float a    = gamma[o] * rsqrtf(var + EPS_);
    ab[o]      = a;
    ab[64 + o] = beta[o] - mean * a;   // conv bias cancels under BN
}

// ---------------- finalize: out[b,o,n] = relu(a * (a>=0 ? max : min) + shift)
__global__ void final_k(const float* __restrict__ ymax,
                        const float* __restrict__ ymin,
                        const float* __restrict__ ab,
                        float* __restrict__ out) {
    int g  = blockIdx.x * 256 + threadIdx.x;  // (b*64+o)*N + n, n fastest
    int n  = g & (N_ - 1);
    int bo = g >> 15;
    int o  = bo & 63;
    int b  = bo >> 6;
    float a  = ab[o];
    float sh = ab[64 + o];
    int src  = (((b << 15) + n) << 6) + o;
    float v  = (a >= 0.f) ? ymax[src] : ymin[src];
    out[g]   = fmaxf(fmaf(a, v, sh), 0.f);
}

extern "C" void kernel_launch(void* const* d_in, const int* in_sizes, int n_in,
                              void* d_out, int out_size, void* d_ws, size_t ws_size,
                              hipStream_t stream) {
    const float* x     = (const float*)d_in[0];
    const int*   eidx  = (const int*)d_in[1];
    const float* W     = (const float*)d_in[2];
    // d_in[3] = conv bias: mathematically absorbed by BatchNorm (training mode)
    const float* gamma = (const float*)d_in[4];
    const float* beta  = (const float*)d_in[5];
    float* out = (float*)d_out;

    char* ws = (char*)d_ws;
    _Float16* xT   = (_Float16*)(ws + 0);          //  8,388,608 B
    _Float16* Wa   = (_Float16*)(ws + 8388608);    //      8,192 B
    _Float16* Wb   = (_Float16*)(ws + 8396800);    //      8,192 B
    float*    ymax = (float*)   (ws + 8404992);    // 16,777,216 B
    float*    ymin = (float*)   (ws + 25182208);   // 16,777,216 B
    float*    part = (float*)   (ws + 41959424);   //    262,144 B
    float*    ab   = (float*)   (ws + 42221568);   //        512 B

    prep_w  <<<16,    256, 0, stream>>>(W, Wa, Wb);
    prep_x  <<<1024,  256, 0, stream>>>(x, xT);
    edge_main<<<512,  256, 0, stream>>>(xT, Wa, Wb, eidx, ymax, ymin, part);
    stats_k <<<1,     64,  0, stream>>>(part, gamma, beta, ab, 512);
    final_k <<<16384, 256, 0, stream>>>(ymax, ymin, ab, out);
}
